// HGNNLayer_3049426780188
// MI455X (gfx1250) — compile-verified, matrix operand order checked by reading, not run on hardware
//
#include <hip/hip_runtime.h>

// ---------------------------------------------------------------------------
// HGNN layer for MI455X (gfx1250, wave32, WMMA).
//   out = x @ WC^T + bC + sum_t scatter_add(dst_t, (1/cnt_t[dst]) * x[src_t] @ WA_t)
// GEMMs: v_wmma_f32_16x16x32_bf16 (f32 accumulate).
// A-tiles are gathered global->LDS with the gfx1250 async-LDS path (ASYNCcnt);
// W stays in registers as B-fragments; x (51.2 MB) is L2-resident (192 MB L2)
// so gathers + scatter atomics are L2 traffic.
// ---------------------------------------------------------------------------

typedef __bf16 bf16_t;
typedef __attribute__((ext_vector_type(8)))  bf16_t v8bf;
typedef __attribute__((ext_vector_type(16))) bf16_t v16bf;
typedef __attribute__((ext_vector_type(8)))  float  v8f;

#define SHUF16(lo, hi) __builtin_shufflevector((lo), (hi), \
    0, 1, 2, 3, 4, 5, 6, 7, 8, 9, 10, 11, 12, 13, 14, 15)

#if __has_builtin(__builtin_amdgcn_global_load_async_to_lds_b128) && \
    __has_builtin(__builtin_amdgcn_s_wait_asynccnt)
#define USE_ASYNC_LDS 1
#else
#define USE_ASYNC_LDS 0
#endif

// Pointee/addrspace types matching the builtin's expected parameters:
// 'int __attribute__((vector_size(16)))' in global (1) / LDS (3) space.
typedef int v4i_vs __attribute__((vector_size(16)));
typedef __attribute__((address_space(1))) v4i_vs* gptr_v4i;
typedef __attribute__((address_space(3))) v4i_vs* lptr_v4i;

// Copy 16B global -> LDS. Async path: no VGPR staging, tracked by ASYNCcnt.
__device__ __forceinline__ void copy16_g2l(const unsigned short* g,
                                           unsigned short* l) {
#if USE_ASYNC_LDS
  __builtin_amdgcn_global_load_async_to_lds_b128(
      (gptr_v4i)(void*)g, (lptr_v4i)(void*)l, /*offset=*/0, /*cpol=*/0);
#else
  *(uint4*)l = *(const uint4*)g;
#endif
}

__device__ __forceinline__ void async_gather_fence() {
#if USE_ASYNC_LDS
  __builtin_amdgcn_s_wait_asynccnt(0);
#endif
}

__device__ __forceinline__ unsigned short f32_to_bf16_bits(float f) {
  unsigned int u = __float_as_uint(f);
  unsigned int r = 0x7FFFu + ((u >> 16) & 1u);   // round-to-nearest-even
  return (unsigned short)((u + r) >> 16);
}

// -------------------------- small prep kernels -----------------------------

__global__ void zero_i32_kernel(int* __restrict__ p, int n) {
  int i = blockIdx.x * blockDim.x + threadIdx.x;
  if (i < n) p[i] = 0;
}

__global__ void count_dst_kernel(const int* __restrict__ dst,
                                 int* __restrict__ cnt, int n) {
  int i = blockIdx.x * blockDim.x + threadIdx.x;
  if (i < n) atomicAdd(cnt + dst[i], 1);
}

__global__ void cvt_bf16_kernel(const float* __restrict__ s,
                                unsigned short* __restrict__ d, int n) {
  int i = blockIdx.x * blockDim.x + threadIdx.x;
  if (i < n) d[i] = f32_to_bf16_bits(s[i]);
}

// W row-major [K][128]  ->  Wt N-major [128][K] (bf16): a wave's B-columns
// become contiguous in K and load as 16B chunks matching the WMMA B layout.
__global__ void cvt_transpose_bf16_kernel(const float* __restrict__ W,
                                          unsigned short* __restrict__ Wt,
                                          int K) {
  int i = blockIdx.x * blockDim.x + threadIdx.x;
  if (i < K * 128) {
    int k = i >> 7;
    int n = i & 127;
    Wt[n * K + k] = f32_to_bf16_bits(W[i]);
  }
}

// ------------------------ dense: out = x @ WC^T + bC -----------------------

__global__ __launch_bounds__(256)
void hgnn_dense_kernel(const unsigned short* __restrict__ xb,   // [N][128] bf16
                       const unsigned short* __restrict__ WCt,  // [128][128] N-major
                       const float* __restrict__ bC,
                       float* __restrict__ out,                 // [N][128]
                       int n_nodes) {
  constexpr int K = 128, KP = K + 8, NFRAG = 4;
  __shared__ __align__(16) unsigned short ldsA[16 * KP];

  const int tid  = threadIdx.x;
  const int wave = tid >> 5;
  const int lane = tid & 31;
  const int col  = lane & 15;                 // N (B/C) or M (A) within tile
  const int klo  = (lane < 16) ? 0 : 8;       // 16-bit operand lane K-offset
  const int n0   = wave * 16;                 // this wave's output columns
  const int mbase = (lane < 16) ? 0 : 8;      // C layout: VGPR v -> row v+mbase

  // Preload B fragments (WC^T columns n0..n0+15) into registers.
  v16bf bfrag[NFRAG];
  {
    const unsigned short* wrow = WCt + (size_t)(n0 + col) * K + klo;
#pragma unroll
    for (int f = 0; f < NFRAG; ++f) {
      v8bf blo = *(const v8bf*)(wrow + f * 32);
      v8bf bhi = *(const v8bf*)(wrow + f * 32 + 16);
      bfrag[f] = SHUF16(blo, bhi);
    }
  }
  const float bias = bC[n0 + col];

  const int n_tiles = (n_nodes + 15) >> 4;
  for (int tile = blockIdx.x; tile < n_tiles; tile += gridDim.x) {
    const int r0 = tile * 16;
    {
      const int r = tid >> 4, c = tid & 15;   // 16 rows x 16 chunks of 16B
      const int node = r0 + r;
      const int ng = (node < n_nodes) ? node : 0;
      copy16_g2l(xb + (size_t)ng * 128 + c * 8, &ldsA[r * KP + c * 8]);
    }
    async_gather_fence();
    __syncthreads();

    v8f acc = {};
    const unsigned short* arow = &ldsA[col * KP + klo];
#pragma unroll
    for (int f = 0; f < NFRAG; ++f) {
      v8bf alo = *(const v8bf*)(arow + f * 32);
      v8bf ahi = *(const v8bf*)(arow + f * 32 + 16);
      v16bf a = SHUF16(alo, ahi);
      acc = __builtin_amdgcn_wmma_f32_16x16x32_bf16(
          false, a, false, bfrag[f], (short)0, acc, false, false);
    }

    // Single per-lane base pointer; rows are a fixed 512B stride apart so the
    // 8 stores become global_store_b32 with constant immediate offsets.
    float* po = out + (size_t)(r0 + mbase) * 128 + n0 + col;
    if (r0 + 16 <= n_nodes) {
#pragma unroll
      for (int v = 0; v < 8; ++v) po[(size_t)v * 128] = acc[v] + bias;
    } else {
#pragma unroll
      for (int v = 0; v < 8; ++v) {
        if (r0 + mbase + v < n_nodes) po[(size_t)v * 128] = acc[v] + bias;
      }
    }
    __syncthreads();
  }
}

// ---------------- per-edge-type gather-GEMM-scatter (WMMA) -----------------

template <int ARITY>
__global__ __launch_bounds__(256)
void hgnn_edge_kernel(const unsigned short* __restrict__ xb,  // [N][128] bf16
                      const unsigned short* __restrict__ Wt,  // [128][ARITY*128] N-major
                      const int* __restrict__ src,            // [E*ARITY]
                      const int* __restrict__ dst,            // [E]
                      const int* __restrict__ cnt,            // [N]
                      float* __restrict__ out,                // [N][128] (+=)
                      int n_edges) {
  constexpr int K     = ARITY * 128;
  constexpr int KP    = K + 8;
  constexpr int NFRAG = ARITY * 4;

  __shared__ __align__(16) unsigned short ldsA[16 * KP];
  __shared__ int   sdst[16];
  __shared__ float snorm[16];

  const int tid  = threadIdx.x;
  const int wave = tid >> 5;
  const int lane = tid & 31;
  const int col  = lane & 15;
  const int klo  = (lane < 16) ? 0 : 8;
  const int n0   = wave * 16;
  const int mbase = (lane < 16) ? 0 : 8;

  // W^T columns for this wave live in registers for the whole kernel.
  v16bf bfrag[NFRAG];
  {
    const unsigned short* wrow = Wt + (size_t)(n0 + col) * K + klo;
#pragma unroll
    for (int f = 0; f < NFRAG; ++f) {
      v8bf blo = *(const v8bf*)(wrow + f * 32);
      v8bf bhi = *(const v8bf*)(wrow + f * 32 + 16);
      bfrag[f] = SHUF16(blo, bhi);
    }
  }

  float* const outc = out + n0 + col;   // column-fixed base for the scatter

  const int n_tiles = (n_edges + 15) >> 4;
  for (int tile = blockIdx.x; tile < n_tiles; tile += gridDim.x) {
    const int e0 = tile * 16;

    if (tid < 16) {
      const int e = e0 + tid;
      const int d = dst[(e < n_edges) ? e : 0];
      sdst[tid]  = d;
      snorm[tid] = (e < n_edges) ? (1.0f / (float)cnt[d]) : 0.0f;
    }
    {
      // Gather 16 rows of [x[src_j] for j<ARITY] straight into LDS (async).
      const int r = tid >> 4, c = tid & 15;
      const int e = e0 + r;
      const int eg = (e < n_edges) ? e : 0;
#pragma unroll
      for (int j = 0; j < ARITY; ++j) {
        const int node = src[eg * ARITY + j];
        copy16_g2l(xb + (size_t)node * 128 + c * 8,
                   &ldsA[r * KP + j * 128 + c * 8]);
      }
    }
    async_gather_fence();
    __syncthreads();

    v8f acc = {};
    const unsigned short* arow = &ldsA[col * KP + klo];
#pragma unroll
    for (int f = 0; f < NFRAG; ++f) {
      v8bf alo = *(const v8bf*)(arow + f * 32);
      v8bf ahi = *(const v8bf*)(arow + f * 32 + 16);
      v16bf a = SHUF16(alo, ahi);
      acc = __builtin_amdgcn_wmma_f32_16x16x32_bf16(
          false, a, false, bfrag[f], (short)0, acc, false, false);
    }

    // Normalize per destination and scatter-add (L2-resident f32 atomics).
    // Tail rows have snorm==0, so the adds are no-ops: no exec juggling.
#pragma unroll
    for (int v = 0; v < 8; ++v) {
      const int m = mbase + v;
      atomicAdd(outc + (size_t)sdst[m] * 128, acc[v] * snorm[m]);
    }
    __syncthreads();
  }
}

// ------------------------------- launcher ----------------------------------

extern "C" void kernel_launch(void* const* d_in, const int* in_sizes, int n_in,
                              void* d_out, int out_size, void* d_ws, size_t ws_size,
                              hipStream_t stream) {
  const float* x    = (const float*)d_in[0];
  const int*   src0 = (const int*)d_in[1];
  const int*   dst0 = (const int*)d_in[2];
  const int*   src1 = (const int*)d_in[3];
  const int*   dst1 = (const int*)d_in[4];
  const int*   src2 = (const int*)d_in[5];
  const int*   dst2 = (const int*)d_in[6];
  const float* WA0  = (const float*)d_in[7];   // [128,128]
  const float* WA1  = (const float*)d_in[8];   // [256,128]
  const float* WA2  = (const float*)d_in[9];   // [384,128]
  const float* WC   = (const float*)d_in[10];  // [128,128] (D_OUT, D_IN)
  const float* bC   = (const float*)d_in[11];  // [128]
  float* out = (float*)d_out;

  const int N  = in_sizes[0] / 128;
  const int E0 = in_sizes[2];
  const int E1 = in_sizes[4];
  const int E2 = in_sizes[6];

  char* ws = (char*)d_ws;
  size_t off = 0;
  auto carve = [&](size_t bytes) -> void* {
    void* p = (void*)(ws + off);
    off = (off + bytes + 255) & ~(size_t)255;
    return p;
  };
  unsigned short* xb  = (unsigned short*)carve((size_t)N * 128 * 2);  // x in bf16
  unsigned short* W0t = (unsigned short*)carve(128 * 128 * 2);
  unsigned short* W1t = (unsigned short*)carve(256 * 128 * 2);
  unsigned short* W2t = (unsigned short*)carve(384 * 128 * 2);
  unsigned short* WCt = (unsigned short*)carve(128 * 128 * 2);        // already N-major
  int* cnt  = (int*)carve((size_t)3 * N * sizeof(int));
  int* cnt0 = cnt, *cnt1 = cnt + N, *cnt2 = cnt + 2 * N;

  const int TPB = 256;
  auto blocks = [&](int n) { return (n + TPB - 1) / TPB; };

  // Per-destination edge counts (zeroed every call: deterministic).
  zero_i32_kernel<<<blocks(3 * N), TPB, 0, stream>>>(cnt, 3 * N);
  count_dst_kernel<<<blocks(E0), TPB, 0, stream>>>(dst0, cnt0, E0);
  count_dst_kernel<<<blocks(E1), TPB, 0, stream>>>(dst1, cnt1, E1);
  count_dst_kernel<<<blocks(E2), TPB, 0, stream>>>(dst2, cnt2, E2);

  // bf16 staging of activations and weights.
  cvt_bf16_kernel<<<blocks(N * 128), TPB, 0, stream>>>(x, xb, N * 128);
  cvt_bf16_kernel<<<blocks(128 * 128), TPB, 0, stream>>>(WC, WCt, 128 * 128);
  cvt_transpose_bf16_kernel<<<blocks(128 * 128), TPB, 0, stream>>>(WA0, W0t, 128);
  cvt_transpose_bf16_kernel<<<blocks(256 * 128), TPB, 0, stream>>>(WA1, W1t, 256);
  cvt_transpose_bf16_kernel<<<blocks(384 * 128), TPB, 0, stream>>>(WA2, W2t, 384);

  // Dense projection first (plain stores initialize d_out)...
  const int dense_tiles = (N + 15) / 16;
  const int dgrid = dense_tiles < 4096 ? dense_tiles : 4096;
  hgnn_dense_kernel<<<dgrid, TPB, 0, stream>>>(xb, WCt, bC, out, N);

  // ...then the three gather-GEMM-scatter passes accumulate on top.
  auto egrid = [](int e) { int t = (e + 15) / 16; return t < 2048 ? t : 2048; };
  hgnn_edge_kernel<1><<<egrid(E0), TPB, 0, stream>>>(xb, W0t, src0, dst0, cnt0, out, E0);
  hgnn_edge_kernel<2><<<egrid(E1), TPB, 0, stream>>>(xb, W1t, src1, dst1, cnt1, out, E1);
  hgnn_edge_kernel<3><<<egrid(E2), TPB, 0, stream>>>(xb, W2t, src2, dst2, cnt2, out, E2);
}
